// COLoRALinear_83382495084801
// MI455X (gfx1250) — matrix operand order repaired and checked by reading, not compiled, not run on hardware
//
#include <hip/hip_runtime.h>
#include <hip/hip_bf16.h>

// ---------------------------------------------------------------------------
// COLoRA fused linear for MI455X (gfx1250, wave32, WMMA).
//
// Strategy:
//  1) Fold base W + shared LoRA + per-task expert LoRA into per-task
//     W_eff[t] (8 x 1024 x 1024 f32 in d_ws)  -- tiny cost (~0.27 GFLOP).
//  2) One big GEMM  out[M=32768, N=1024] = x[M,K=1024] @ W_eff[task]^T + b
//     using split-bf16 (hi/lo) 3-term products with f32 accumulation on
//     v_wmma_f32_16x16x32_bf16 for near-fp32 accuracy at bf16 matrix rate.
// ---------------------------------------------------------------------------

typedef __attribute__((ext_vector_type(16))) __bf16 v16bf;
typedef __attribute__((ext_vector_type(8)))  __bf16 bf16x8;
typedef __attribute__((ext_vector_type(4)))  __bf16 bf16x4;
typedef __attribute__((ext_vector_type(8)))  float  v8f;

#define D_IN   1024
#define D_OUT  1024
#define RLORA  8
#define NTASK  8
#define MTOT   32768      // B * S = 8 * 4096
#define BM     128
#define BN     128
#define BK     32
#define LDSS   40         // padded LDS row stride (bf16 elems): 80B -> conflict-free

// ---------------------------------------------------------------------------
// Kernel 1: build W_eff[t][o][i] = W + c1*sharedB@sharedA + c2*expB[t]@expA[t]
// ---------------------------------------------------------------------------
__global__ __launch_bounds__(256)
void colora_build_weff(const float* __restrict__ W,
                       const float* __restrict__ sA,   // [R, D_IN]
                       const float* __restrict__ sB,   // [D_OUT, R]
                       const float* __restrict__ eA,   // [T, R, D_IN]
                       const float* __restrict__ eB,   // [T, D_OUT, R]
                       const float* __restrict__ collab,
                       float* __restrict__ weff)       // [T, D_OUT, D_IN]
{
  int idx = blockIdx.x * 256 + threadIdx.x;           // 0 .. 8M-1
  int t   = idx >> 20;                                // task
  int rem = idx & ((1 << 20) - 1);
  int o   = rem >> 10;
  int i   = rem & 1023;

  float cw = 1.0f / (1.0f + expf(-collab[0]));
  float c1 = cw * 2.0f;                               // SCALING = alpha/r = 2
  float c2 = (1.0f - cw) * 2.0f;

  float s = 0.0f, e = 0.0f;
#pragma unroll
  for (int r = 0; r < RLORA; ++r) {
    s += sB[o * RLORA + r] * sA[r * D_IN + i];
    e += eB[(t * D_OUT + o) * RLORA + r] * eA[(t * RLORA + r) * D_IN + i];
  }
  weff[idx] = W[rem] + c1 * s + c2 * e;
}

// ---------------------------------------------------------------------------
// Kernel 2: GEMM with split-bf16 WMMA.
//   grid = (N/BN, M/BM), block = 256 (8 waves).
//   Wave grid 4(M) x 2(N): each wave -> 32x64 outputs = 2x4 tiles of 16x16.
// ---------------------------------------------------------------------------
__global__ __launch_bounds__(256)
void colora_gemm(const float* __restrict__ x,
                 const int*   __restrict__ task_ids,
                 const float* __restrict__ weff,
                 const float* __restrict__ bias,
                 float* __restrict__ out)
{
  __shared__ __bf16 Ahi[BM * LDSS];
  __shared__ __bf16 Alo[BM * LDSS];
  __shared__ __bf16 Bhi[BN * LDSS];
  __shared__ __bf16 Blo[BN * LDSS];

  const int tid   = threadIdx.x;
  const int lane  = tid & 31;
  const int wave  = tid >> 5;
  const int waveM = wave & 3;        // 0..3 -> 32-row slab
  const int waveN = wave >> 2;       // 0..1 -> 64-col slab
  const int l16   = lane & 15;
  const int lh    = lane >> 4;       // half-wave selector

  const int m0 = blockIdx.y * BM;
  const int n0 = blockIdx.x * BN;
  const int task = task_ids[m0 >> 12];               // 4096 rows per batch elem
  const float* Wt = weff + (long)task * D_OUT * D_IN;

  v8f acc[8];
#pragma unroll
  for (int t = 0; t < 8; ++t)
#pragma unroll
    for (int j = 0; j < 8; ++j) acc[t][j] = 0.0f;

  for (int kk = 0; kk < D_IN / BK; ++kk) {
    const int k0 = kk * BK;
    __syncthreads();

    // ---- stage tiles into LDS, splitting f32 -> bf16 hi + bf16 lo --------
#pragma unroll
    for (int u = 0; u < 4; ++u) {
      const int f   = tid + u * 256;      // 0..1023 float4 slots
      const int row = f >> 3;             // 0..127
      const int kc  = (f & 7) << 2;       // 0,4,...,28

      const float4 va = *(const float4*)(x  + (long)(m0 + row) * D_IN + k0 + kc);
      const float4 vb = *(const float4*)(Wt + (long)(n0 + row) * D_IN + k0 + kc);
      const float av[4] = {va.x, va.y, va.z, va.w};
      const float bv[4] = {vb.x, vb.y, vb.z, vb.w};
      bf16x4 ah, al, bh, bl;
#pragma unroll
      for (int j = 0; j < 4; ++j) {
        __bf16 h = (__bf16)av[j];
        ah[j] = h;  al[j] = (__bf16)(av[j] - (float)h);
        __bf16 g = (__bf16)bv[j];
        bh[j] = g;  bl[j] = (__bf16)(bv[j] - (float)g);
      }
      *(bf16x4*)&Ahi[row * LDSS + kc] = ah;
      *(bf16x4*)&Alo[row * LDSS + kc] = al;
      *(bf16x4*)&Bhi[row * LDSS + kc] = bh;
      *(bf16x4*)&Blo[row * LDSS + kc] = bl;
    }

    // prefetch next K-tile while we compute (global_prefetch_b8)
    if (kk + 1 < D_IN / BK) {
      const int row = tid >> 3, kc = (tid & 7) << 2;
      __builtin_prefetch(x  + (long)(m0 + row) * D_IN + k0 + BK + kc, 0, 1);
      __builtin_prefetch(Wt + (long)(n0 + row) * D_IN + k0 + BK + kc, 0, 1);
    }
    __syncthreads();

    // ---- load A fragments (16x32 bf16, ISA layout) -----------------------
    // lanes 0-15 : elems 0..7 = K0..7,  elems 8..15 = K16..23  (row M = l16)
    // lanes 16-31: elems 0..7 = K8..15, elems 8..15 = K24..31
    v16bf aH[2], aL[2];
#pragma unroll
    for (int mt = 0; mt < 2; ++mt) {
      const int row = waveM * 32 + mt * 16 + l16;
      const __bf16* pH = &Ahi[row * LDSS + lh * 8];
      const __bf16* pL = &Alo[row * LDSS + lh * 8];
      bf16x8 h0 = *(const bf16x8*)pH;
      bf16x8 h1 = *(const bf16x8*)(pH + 16);
      bf16x8 g0 = *(const bf16x8*)pL;
      bf16x8 g1 = *(const bf16x8*)(pL + 16);
#pragma unroll
      for (int j = 0; j < 8; ++j) {
        aH[mt][j] = h0[j]; aH[mt][8 + j] = h1[j];
        aL[mt][j] = g0[j]; aL[mt][8 + j] = g1[j];
      }
    }

    // ---- load B fragments (32x16 bf16): lane = output column, ------------
    // lanes 0-15 hold K0..7,K16..23; lanes 16-31 hold K8..15,K24..31 (same as A)
    v16bf bH[4], bL[4];
#pragma unroll
    for (int nt = 0; nt < 4; ++nt) {
      const int row = waveN * 64 + nt * 16 + l16;
      const __bf16* pH = &Bhi[row * LDSS + lh * 8];
      const __bf16* pL = &Blo[row * LDSS + lh * 8];
      bf16x8 h0 = *(const bf16x8*)pH;
      bf16x8 h1 = *(const bf16x8*)(pH + 16);
      bf16x8 g0 = *(const bf16x8*)pL;
      bf16x8 g1 = *(const bf16x8*)(pL + 16);
#pragma unroll
      for (int j = 0; j < 8; ++j) {
        bH[nt][j] = h0[j]; bH[nt][8 + j] = h1[j];
        bL[nt][j] = g0[j]; bL[nt][8 + j] = g1[j];
      }
    }

    // ---- 3-term split-bf16 WMMA: hi*hi + hi*lo + lo*hi -------------------
#pragma unroll
    for (int nt = 0; nt < 4; ++nt) {
#pragma unroll
      for (int mt = 0; mt < 2; ++mt) {
        const int t = nt * 2 + mt;
        acc[t] = __builtin_amdgcn_wmma_f32_16x16x32_bf16(
            false, aH[mt], false, bH[nt], (short)0, acc[t], false, false);
        acc[t] = __builtin_amdgcn_wmma_f32_16x16x32_bf16(
            false, aH[mt], false, bL[nt], (short)0, acc[t], false, false);
        acc[t] = __builtin_amdgcn_wmma_f32_16x16x32_bf16(
            false, aL[mt], false, bH[nt], (short)0, acc[t], false, false);
      }
    }
  }

  // ---- epilogue: C layout = VGPR j -> (M=j | M=j+8 by lane half), N=l16 ---
#pragma unroll
  for (int nt = 0; nt < 4; ++nt) {
    const int col = n0 + waveN * 64 + nt * 16 + l16;
    const float bv = bias[col];
#pragma unroll
    for (int mt = 0; mt < 2; ++mt) {
      const int t = nt * 2 + mt;
      const int rbase = m0 + waveM * 32 + mt * 16 + lh * 8;
#pragma unroll
      for (int j = 0; j < 8; ++j)
        out[(long)(rbase + j) * D_OUT + col] = acc[t][j] + bv;
    }
  }
}

// ---------------------------------------------------------------------------
extern "C" void kernel_launch(void* const* d_in, const int* in_sizes, int n_in,
                              void* d_out, int out_size, void* d_ws, size_t ws_size,
                              hipStream_t stream) {
  (void)in_sizes; (void)n_in; (void)out_size; (void)ws_size;
  const float* x   = (const float*)d_in[0];   // [8,4096,1024]
  const int*   tid = (const int*)  d_in[1];   // [8]
  const float* W   = (const float*)d_in[2];   // [1024,1024]
  const float* b   = (const float*)d_in[3];   // [1024]
  const float* sA  = (const float*)d_in[4];   // [8,1024]
  const float* sB  = (const float*)d_in[5];   // [1024,8]
  const float* eA  = (const float*)d_in[6];   // [8,8,1024]
  const float* eB  = (const float*)d_in[7];   // [8,1024,8]
  const float* cw  = (const float*)d_in[8];   // scalar
  float* out  = (float*)d_out;                // [8,4096,1024]
  float* weff = (float*)d_ws;                 // 8*1024*1024 f32 = 32 MB scratch

  colora_build_weff<<<(NTASK * D_OUT * D_IN) / 256, 256, 0, stream>>>(
      W, sA, sB, eA, eB, cw, weff);

  dim3 grid(D_OUT / BN, MTOT / BM);            // (8, 256)
  colora_gemm<<<grid, 256, 0, stream>>>(x, tid, weff, b, out);
}
